// AttentionWithCAE_26706106647171
// MI455X (gfx1250) — matrix operand order repaired and checked by reading, not compiled, not run on hardware
//
#include <hip/hip_runtime.h>

typedef __bf16 bf16;
typedef __attribute__((ext_vector_type(16))) __bf16 v16bf;
typedef __attribute__((ext_vector_type(8)))  __bf16 v8bf;
typedef __attribute__((ext_vector_type(8)))  float  v8f;

#define BATCH 8
#define SEQ   1024
#define DIM   768
#define NH    12
#define HD    64
#define DIM3  2304
#define SCALE 0.125f

// ---------------------------------------------------------------------------
// WMMA helpers (CDNA5 wave32, V_WMMA_F32_16X16X32_BF16)
// ---------------------------------------------------------------------------
__device__ __forceinline__ v8f wmma_bf16(v16bf a, v16bf b, v8f c) {
  return __builtin_amdgcn_wmma_f32_16x16x32_bf16(false, a, false, b,
                                                 (short)0, c, false, false);
}

// A-fragment: 16x32 bf16 tile, row-major source, ld in elements.
// lane%16 = row M; lanes 0-15 hold K {0..7,16..23}, lanes 16-31 hold {8..15,24..31}.
__device__ __forceinline__ v16bf load_frag_a(const bf16* tile, int ld, int lane) {
  const bf16* p = tile + (lane & 15) * ld + ((lane & 16) ? 8 : 0);
  v8bf lo = *(const v8bf*)(p);
  v8bf hi = *(const v8bf*)(p + 16);
  return __builtin_shufflevector(lo, hi, 0,1,2,3,4,5,6,7,8,9,10,11,12,13,14,15);
}

// B-fragment: 32x16 bf16 tile, row-major source (K rows), ld in elements.
// lane%16 = col N; lanes 0-15 hold K {0..7,16..23}, lanes 16-31 hold {8..15,24..31}.
__device__ __forceinline__ v16bf load_frag_b(const bf16* tile, int ld, int lane) {
  const bf16* p = tile + (lane & 15) + ((lane & 16) ? 8 : 0) * ld;
  v16bf r;
#pragma unroll
  for (int s = 0; s < 8; ++s) {
    r[s] = p[s * ld];
    r[s + 8] = p[(16 + s) * ld];
  }
  return r;
}

// ---------------------------------------------------------------------------
// Pack kernels: fp32 -> bf16 (and weight transposes so B-frags are contiguous)
// ---------------------------------------------------------------------------
__global__ void cvt_f32_bf16_kernel(const float* __restrict__ in,
                                    bf16* __restrict__ out, int n) {
  int i = blockIdx.x * 256 + threadIdx.x;
  if (i < n) out[i] = (bf16)in[i];
}

// in: [Nn][Kk] row-major fp32, out: [Kk][Nn] bf16 (out[k*Nn+n] = in[n*Kk+k])
__global__ void transpose_f32_bf16_kernel(const float* __restrict__ in,
                                          bf16* __restrict__ out,
                                          int Nn, int Kk) {
  int i = blockIdx.x * 256 + threadIdx.x;
  if (i >= Nn * Kk) return;
  int k = i / Nn;
  int n = i - k * Nn;
  out[i] = (bf16)in[n * Kk + k];
}

// ---------------------------------------------------------------------------
// QKV GEMM: [8192,768]bf16 @ [768,2304]bf16 -> split into q/kT/v (bf16)
// 1 wave per 16x64 output tile (4 accumulators share one A-frag).
// Latency hiding via zero-register-cost global_prefetch_b8 of the next
// k-iteration's A/B rows (register double-buffering spilled; prefetch doesn't).
// Epilogue: q scaled by 1/sqrt(hd) + q_bias, k transposed to [B,H,64,N], v + v_bias.
// ---------------------------------------------------------------------------
__global__ void qkv_gemm_kernel(const bf16* __restrict__ X,
                                const bf16* __restrict__ Wt,   // [768][2304]
                                const float* __restrict__ q_bias,
                                const float* __restrict__ v_bias,
                                bf16* __restrict__ q,    // [B,H,N,64]
                                bf16* __restrict__ kT,   // [B,H,64,N]
                                bf16* __restrict__ v) {  // [B,H,N,64]
  const int NT = DIM3 / 64;  // 36
  int wave = blockIdx.x * 8 + (threadIdx.x >> 5);
  int lane = threadIdx.x & 31;
  int mt = wave / NT;
  int nt = wave - mt * NT;
  int nbase = nt * 64;

  const bf16* arow = X + (size_t)mt * 16 * DIM;
  v8f zf = {};
  v8f acc[4] = {zf, zf, zf, zf};

  for (int k0 = 0; k0 < DIM; k0 += 32) {
    // warm next iteration's rows in the WGP cache (no VGPR cost)
    __builtin_prefetch(arow + (size_t)(lane & 15) * DIM + k0 + 32, 0, 0);
    __builtin_prefetch(Wt + (size_t)(k0 + 32 + lane) * DIM3 + nbase, 0, 0);

    v16bf a = load_frag_a(arow + k0, DIM, lane);
#pragma unroll
    for (int t = 0; t < 4; ++t) {
      v16bf b = load_frag_b(Wt + (size_t)k0 * DIM3 + nbase + t * 16, DIM3, lane);
      acc[t] = wmma_bf16(a, b, acc[t]);
    }
  }

  int half8 = (lane & 16) ? 8 : 0;
  int cl = lane & 15;
#pragma unroll
  for (int t = 0; t < 4; ++t) {
    int n = nbase + t * 16 + cl;
#pragma unroll
    for (int i = 0; i < 8; ++i) {
      int m = mt * 16 + i + half8;
      int b = m >> 10, row = m & 1023;
      float val = acc[t][i];
      if (n < DIM) {                       // Q: bias + scale
        int hh = n >> 6, d = n & 63;
        q[(((size_t)b * NH + hh) * SEQ + row) * HD + d] =
            (bf16)((val + q_bias[n]) * SCALE);
      } else if (n < 2 * DIM) {            // K: zero bias, store transposed
        int nn = n - DIM;
        int hh = nn >> 6, d = nn & 63;
        kT[(((size_t)b * NH + hh) * HD + d) * SEQ + row] = (bf16)val;
      } else {                             // V: bias
        int nn = n - 2 * DIM;
        int hh = nn >> 6, d = nn & 63;
        v[(((size_t)b * NH + hh) * SEQ + row) * HD + d] =
            (bf16)(val + v_bias[nn]);
      }
    }
  }
}

// ---------------------------------------------------------------------------
// Flash attention: 1 wave per (b, h, 16-query tile); online softmax over
// 32-key groups; P routed through per-wave LDS slice to reach A-frag layout.
// Next key-group K/V rows are warmed with global_prefetch_b8.
// Output written bf16 [B, N, H*64] (row-major) for the proj GEMM.
// ---------------------------------------------------------------------------
#define PLD 40  // padded LDS row stride (bf16 elems) to dodge bank conflicts

__global__ void attn_kernel(const bf16* __restrict__ q,
                            const bf16* __restrict__ kT,
                            const bf16* __restrict__ v,
                            const int* __restrict__ mask,   // [B,N], nonzero = masked
                            bf16* __restrict__ attn_out) {  // [B,N,768] bf16
  __shared__ bf16 ldsP[8][16 * PLD];
  int wave = threadIdx.x >> 5;
  int lane = threadIdx.x & 31;
  int gw = blockIdx.x * 8 + wave;
  int qt = gw & 63;          // 64 query tiles of 16
  int bh = gw >> 6;
  int h = bh % NH, b = bh / NH;

  const bf16* qbase = q + (((size_t)b * NH + h) * SEQ + qt * 16) * HD;
  const bf16* ktb   = kT + ((size_t)b * NH + h) * HD * SEQ;
  const bf16* vb    = v + ((size_t)b * NH + h) * SEQ * HD;
  const int* mrow   = mask + b * SEQ;
  bf16* myP = &ldsP[wave][0];

  // Q fragments (d = 0..31, 32..63), resident for whole pass
  v16bf aq0 = load_frag_a(qbase + 0, HD, lane);
  v16bf aq1 = load_frag_a(qbase + 32, HD, lane);

  float mrun[8], lrun[8];
#pragma unroll
  for (int i = 0; i < 8; ++i) { mrun[i] = -1e30f; lrun[i] = 0.f; }
  v8f zf = {};
  v8f o0 = zf, o1 = zf, o2 = zf, o3 = zf;

  for (int kg = 0; kg < SEQ / 32; ++kg) {
    int key0 = kg * 32;
    // ---- scores: S[16q x 32k] = Q[16x64] @ K^T[64x32] (two 16-col halves)
    v8f c0 = zf, c1 = zf;
    c0 = wmma_bf16(aq0, load_frag_b(ktb + 0 * SEQ + key0, SEQ, lane), c0);
    c0 = wmma_bf16(aq1, load_frag_b(ktb + 32 * SEQ + key0, SEQ, lane), c0);
    c1 = wmma_bf16(aq0, load_frag_b(ktb + 0 * SEQ + key0 + 16, SEQ, lane), c1);
    c1 = wmma_bf16(aq1, load_frag_b(ktb + 32 * SEQ + key0 + 16, SEQ, lane), c1);

    // ---- warm next key-group K rows and next group's V rows (no VGPR cost)
    const bf16* nk = ktb + (size_t)lane * SEQ + key0 + 32;
    __builtin_prefetch(nk, 0, 0);
    __builtin_prefetch(nk + 32 * SEQ, 0, 0);
    __builtin_prefetch(vb + (size_t)(key0 + 32 + lane) * HD, 0, 0);

    // ---- mask + online softmax (rows live across 16-lane halves)
    bool msk0 = mrow[key0 + (lane & 15)] != 0;        // one mask per lane: all 8
    bool msk1 = mrow[key0 + 16 + (lane & 15)] != 0;   // rows share this column
    float mnew[8], rescale[8];
#pragma unroll
    for (int i = 0; i < 8; ++i) {
      float s0 = msk0 ? -3e38f : c0[i];
      float s1 = msk1 ? -3e38f : c1[i];
      c0[i] = s0; c1[i] = s1;
      float mx = fmaxf(s0, s1);
      mx = fmaxf(mx, __shfl_xor(mx, 1, 32));
      mx = fmaxf(mx, __shfl_xor(mx, 2, 32));
      mx = fmaxf(mx, __shfl_xor(mx, 4, 32));
      mx = fmaxf(mx, __shfl_xor(mx, 8, 32));
      mnew[i] = fmaxf(mrun[i], mx);
      rescale[i] = __expf(mrun[i] - mnew[i]);
    }

    // ---- P = exp(S - mnew): write C-layout -> LDS, row sums on the fly
    int mbase = (lane & 16) ? 8 : 0;
    int cl = lane & 15;
#pragma unroll
    for (int i = 0; i < 8; ++i) {
      int m = i + mbase;
      float p0 = __expf(c0[i] - mnew[i]);
      float p1 = __expf(c1[i] - mnew[i]);
      myP[m * PLD + cl] = (bf16)p0;
      myP[m * PLD + 16 + cl] = (bf16)p1;
      float rs = p0 + p1;
      rs += __shfl_xor(rs, 1, 32);
      rs += __shfl_xor(rs, 2, 32);
      rs += __shfl_xor(rs, 4, 32);
      rs += __shfl_xor(rs, 8, 32);
      lrun[i] = lrun[i] * rescale[i] + rs;
      mrun[i] = mnew[i];
    }
    __syncthreads();  // uniform across all 8 waves (equal trip counts)

    // ---- reload P as A-frag [16q x 32k] and accumulate O += P @ V[32x64]
    v16bf ap = load_frag_a(myP, PLD, lane);
#pragma unroll
    for (int i = 0; i < 8; ++i) {
      o0[i] *= rescale[i]; o1[i] *= rescale[i];
      o2[i] *= rescale[i]; o3[i] *= rescale[i];
    }
    const bf16* vt = vb + (size_t)key0 * HD;
    o0 = wmma_bf16(ap, load_frag_b(vt + 0, HD, lane), o0);
    o1 = wmma_bf16(ap, load_frag_b(vt + 16, HD, lane), o1);
    o2 = wmma_bf16(ap, load_frag_b(vt + 32, HD, lane), o2);
    o3 = wmma_bf16(ap, load_frag_b(vt + 48, HD, lane), o3);
  }

  // ---- normalize and store [B,N,768] bf16
  int mbase = (lane & 16) ? 8 : 0;
  int cl = lane & 15;
#pragma unroll
  for (int i = 0; i < 8; ++i) {
    int m = qt * 16 + i + mbase;
    float inv = 1.0f / lrun[i];
    size_t rowbase = ((size_t)b * SEQ + m) * DIM + h * HD;
    attn_out[rowbase + 0 + cl] = (bf16)(o0[i] * inv);
    attn_out[rowbase + 16 + cl] = (bf16)(o1[i] * inv);
    attn_out[rowbase + 32 + cl] = (bf16)(o2[i] * inv);
    attn_out[rowbase + 48 + cl] = (bf16)(o3[i] * inv);
  }
}

// ---------------------------------------------------------------------------
// Proj GEMM: [8192,768]bf16 @ [768,768]bf16 + bias -> fp32 out
// ---------------------------------------------------------------------------
__global__ void proj_gemm_kernel(const bf16* __restrict__ A,
                                 const bf16* __restrict__ Wt,  // [768][768]
                                 const float* __restrict__ bias,
                                 float* __restrict__ out) {
  const int NT = DIM / 64;  // 12
  int wave = blockIdx.x * 8 + (threadIdx.x >> 5);
  int lane = threadIdx.x & 31;
  int mt = wave / NT;
  int nt = wave - mt * NT;
  int nbase = nt * 64;

  const bf16* arow = A + (size_t)mt * 16 * DIM;
  v8f zf = {};
  v8f acc[4] = {zf, zf, zf, zf};

  for (int k0 = 0; k0 < DIM; k0 += 32) {
    __builtin_prefetch(arow + (size_t)(lane & 15) * DIM + k0 + 32, 0, 0);
    __builtin_prefetch(Wt + (size_t)(k0 + 32 + lane) * DIM + nbase, 0, 0);

    v16bf a = load_frag_a(arow + k0, DIM, lane);
#pragma unroll
    for (int t = 0; t < 4; ++t) {
      v16bf b = load_frag_b(Wt + (size_t)k0 * DIM + nbase + t * 16, DIM, lane);
      acc[t] = wmma_bf16(a, b, acc[t]);
    }
  }

  int half8 = (lane & 16) ? 8 : 0;
  int cl = lane & 15;
#pragma unroll
  for (int t = 0; t < 4; ++t) {
    int n = nbase + t * 16 + cl;
    float bb = bias[n];
#pragma unroll
    for (int i = 0; i < 8; ++i) {
      int m = mt * 16 + i + half8;
      out[(size_t)m * DIM + n] = acc[t][i] + bb;
    }
  }
}

// ---------------------------------------------------------------------------
// Host launcher
// ---------------------------------------------------------------------------
extern "C" void kernel_launch(void* const* d_in, const int* in_sizes, int n_in,
                              void* d_out, int out_size, void* d_ws, size_t ws_size,
                              hipStream_t stream) {
  const float* x      = (const float*)d_in[0];
  const int*   mask   = (const int*)d_in[1];
  const float* qkv_w  = (const float*)d_in[2];
  const float* q_bias = (const float*)d_in[3];
  const float* v_bias = (const float*)d_in[4];
  const float* proj_w = (const float*)d_in[5];
  const float* proj_b = (const float*)d_in[6];
  float* out = (float*)d_out;

  char* ws = (char*)d_ws;
  const size_t XBF   = (size_t)BATCH * SEQ * DIM * 2;      // 12,582,912
  const size_t QKVWT = (size_t)DIM * DIM3 * 2;             //  3,538,944
  const size_t PRJWT = (size_t)DIM * DIM * 2;              //  1,179,648
  const size_t HEADS = (size_t)BATCH * NH * SEQ * HD * 2;  // 12,582,912

  bf16* xbf    = (bf16*)(ws);
  bf16* qkvwT  = (bf16*)(ws + XBF);
  bf16* projwT = (bf16*)(ws + XBF + QKVWT);
  bf16* qbuf   = (bf16*)(ws + XBF + QKVWT + PRJWT);
  bf16* kTbuf  = (bf16*)(ws + XBF + QKVWT + PRJWT + HEADS);
  bf16* vbuf   = (bf16*)(ws + XBF + QKVWT + PRJWT + 2 * HEADS);
  bf16* attnbf = (bf16*)(ws + XBF + QKVWT + PRJWT + 3 * HEADS);

  // 1) pack to bf16 (+ K-major weight transposes)
  {
    int n = BATCH * SEQ * DIM;  // 6,291,456
    cvt_f32_bf16_kernel<<<n / 256, 256, 0, stream>>>(x, xbf, n);
  }
  transpose_f32_bf16_kernel<<<(DIM3 * DIM) / 256, 256, 0, stream>>>(
      qkv_w, qkvwT, DIM3, DIM);
  transpose_f32_bf16_kernel<<<(DIM * DIM) / 256, 256, 0, stream>>>(
      proj_w, projwT, DIM, DIM);

  // 2) QKV projection -> q (scaled, biased) / kT (transposed) / v (biased)
  {
    int waves = (BATCH * SEQ / 16) * (DIM3 / 64);  // 512*36 = 18432
    qkv_gemm_kernel<<<waves / 8, 256, 0, stream>>>(xbf, qkvwT, q_bias, v_bias,
                                                   qbuf, kTbuf, vbuf);
  }

  // 3) flash attention, bf16 out in [B,N,768]
  {
    int waves = BATCH * NH * (SEQ / 16);  // 6144
    attn_kernel<<<waves / 8, 256, 0, stream>>>(qbuf, kTbuf, vbuf, mask, attnbf);
  }

  // 4) output projection (fp32 + bias)
  {
    int waves = (BATCH * SEQ / 16) * (DIM / 64);  // 6144
    proj_gemm_kernel<<<waves / 8, 256, 0, stream>>>(attnbf, projwT, proj_b, out);
  }
}
